// Cross_MultiAttention_53188874994185
// MI455X (gfx1250) — compile-verified, matrix operand order checked by reading, not run on hardware
//
#include <hip/hip_runtime.h>
#include <hip/hip_bf16.h>

// Cross multi-head attention, bf16 WMMA pipeline for gfx1250 (MI455X).
// B=4, NQ=NKV=1024, D=1024, H=16, DEPTH=64.  All matmuls on
// v_wmma_f32_16x16x32_bf16 (f32 accumulate); softmax in f32.
// Projection GEMMs use double-buffered GLOBAL_LOAD_ASYNC_TO_LDS staging
// (ASYNCcnt) when the toolchain exposes the builtin; sync fallback otherwise.

typedef __bf16 bf16;
typedef __attribute__((ext_vector_type(8)))  __bf16 v8bf;
typedef __attribute__((ext_vector_type(16))) __bf16 v16bf;
typedef __attribute__((ext_vector_type(8)))  float  v8f;
typedef __attribute__((ext_vector_type(4)))  int    v4i;

#define BB    4
#define NQS   1024
#define NKVS  1024
#define DDIM  1024
#define NH    16
#define HDEP  64
#define ATT_SCALE 0.03125f   // 1024^-0.5

#if defined(__gfx1250__) && __has_builtin(__builtin_amdgcn_global_load_async_to_lds_b128)
#define USE_ASYNC 1
#else
#define USE_ASYNC 0
#endif

static __device__ __forceinline__ void wait_async0() {
#if __has_builtin(__builtin_amdgcn_s_wait_asynccnt)
  __builtin_amdgcn_s_wait_asynccnt(0);
#else
  asm volatile("s_wait_asynccnt 0x0" ::: "memory");
#endif
}

#if USE_ASYNC
// 16B global -> LDS async copy (tracked by ASYNCcnt, not LOADcnt).
// Builtin signature (from hipcc diagnostic): arg1 = AS(1) v4i* global src.
typedef __attribute__((address_space(1))) v4i gv4i_t;
typedef __attribute__((address_space(3))) v4i lv4i_t;
static __device__ __forceinline__ void async_cp16(bf16* lds, const bf16* glb) {
  __builtin_amdgcn_global_load_async_to_lds_b128(
      (gv4i_t*)glb, (lv4i_t*)lds, 0, 0);
}
#endif

static __device__ __forceinline__ v16bf make_frag(v8bf lo, v8bf hi) {
  return __builtin_shufflevector(lo, hi, 0,1,2,3,4,5,6,7,8,9,10,11,12,13,14,15);
}

static __device__ __forceinline__ v8bf cvt8(float4 a, float4 b) {
  v8bf r;
  r[0] = (bf16)a.x; r[1] = (bf16)a.y; r[2] = (bf16)a.z; r[3] = (bf16)a.w;
  r[4] = (bf16)b.x; r[5] = (bf16)b.y; r[6] = (bf16)b.z; r[7] = (bf16)b.w;
  return r;
}

static __device__ __forceinline__ v8f vzero8() {
  v8f z = {0.f, 0.f, 0.f, 0.f, 0.f, 0.f, 0.f, 0.f};
  return z;
}

// ---------------------------------------------------------------- convert
__global__ __launch_bounds__(256) void cvt_f32_bf16_kernel(
    const float* __restrict__ s, bf16* __restrict__ d, int n8) {
  int i = blockIdx.x * 256 + threadIdx.x;
  if (i >= n8) return;
  const float4* p = (const float4*)s + (size_t)i * 2;
  float4 a = p[0], b = p[1];
  *((v8bf*)d + i) = cvt8(a, b);
}

// --------------------------------------------------- C = A @ W^T + bias
// A: [M,K] bf16 row-major, W: [N,K] bf16 row-major (torch Linear weight).
// Block tile 128x128, k-step 32, 8 waves: each wave 64x32 (4x2 WMMA tiles).
template <bool OUT_BF16>
__global__ __launch_bounds__(256) void gemm_xt_bias(
    const bf16* __restrict__ A, const bf16* __restrict__ W,
    const float* __restrict__ bias,
    bf16* __restrict__ Cbf, float* __restrict__ Cf,
    int M, int N, int K) {
#if USE_ASYNC
  __shared__ bf16 As[2][128 * 40];  // stride 40 bf16 = 80B, 16B aligned, skewed
  __shared__ bf16 Bs[2][128 * 40];
#else
  __shared__ bf16 As[1][128 * 40];
  __shared__ bf16 Bs[1][128 * 40];
#endif

  const int t     = threadIdx.x;
  const int lane  = t & 31;
  const int w     = t >> 5;
  const int m0    = blockIdx.y * 128;
  const int n0    = blockIdx.x * 128;
  const int waveM = (w >> 2) * 64;
  const int waveN = (w & 3) * 32;
  const int mLane = lane & 15;
  const int kOff  = (lane >> 4) * 8;   // K-half select per documented 16-bit layout

  // staging coordinates: 512 uint4 per operand tile, 2 per thread
  const int i0 = t, i1 = t + 256;
  const int r0 = i0 >> 2, c0 = (i0 & 3) * 8;
  const int r1 = i1 >> 2, c1 = (i1 & 3) * 8;
  const bf16* A0 = A + (size_t)(m0 + r0) * K + c0;
  const bf16* A1 = A + (size_t)(m0 + r1) * K + c1;
  const bf16* W0 = W + (size_t)(n0 + r0) * K + c0;
  const bf16* W1 = W + (size_t)(n0 + r1) * K + c1;

  v8f acc[4][2];
#pragma unroll
  for (int r = 0; r < 4; ++r)
#pragma unroll
    for (int c = 0; c < 2; ++c) acc[r][c] = vzero8();

#if USE_ASYNC
  // prologue: stage tile 0 into buffer 0
  async_cp16(&As[0][r0 * 40 + c0], A0);
  async_cp16(&As[0][r1 * 40 + c1], A1);
  async_cp16(&Bs[0][r0 * 40 + c0], W0);
  async_cp16(&Bs[0][r1 * 40 + c1], W1);
  wait_async0();
  __syncthreads();

  int buf = 0;
  for (int k0 = 0; k0 < K; k0 += 32) {
    if (k0 + 32 < K) {               // stage next tile into other buffer
      const int nb = buf ^ 1, kn = k0 + 32;
      async_cp16(&As[nb][r0 * 40 + c0], A0 + kn);
      async_cp16(&As[nb][r1 * 40 + c1], A1 + kn);
      async_cp16(&Bs[nb][r0 * 40 + c0], W0 + kn);
      async_cp16(&Bs[nb][r1 * 40 + c1], W1 + kn);
    }

    v16bf afr[4], bfr[2];
#pragma unroll
    for (int r = 0; r < 4; ++r) {
      int idx = (waveM + r * 16 + mLane) * 40 + kOff;
      afr[r] = make_frag(*(const v8bf*)&As[buf][idx], *(const v8bf*)&As[buf][idx + 16]);
    }
#pragma unroll
    for (int c = 0; c < 2; ++c) {
      int idx = (waveN + c * 16 + mLane) * 40 + kOff;
      bfr[c] = make_frag(*(const v8bf*)&Bs[buf][idx], *(const v8bf*)&Bs[buf][idx + 16]);
    }
#pragma unroll
    for (int r = 0; r < 4; ++r)
#pragma unroll
      for (int c = 0; c < 2; ++c)
        acc[r][c] = __builtin_amdgcn_wmma_f32_16x16x32_bf16(
            false, afr[r], false, bfr[c], (short)0, acc[r][c], false, false);

    wait_async0();     // next tile's async copies done
    __syncthreads();   // visible to all waves; buf safe to overwrite next iter
    buf ^= 1;
  }
#else
  for (int k0 = 0; k0 < K; k0 += 32) {
    *(uint4*)(&As[0][r0 * 40 + c0]) = *(const uint4*)(A0 + k0);
    *(uint4*)(&As[0][r1 * 40 + c1]) = *(const uint4*)(A1 + k0);
    *(uint4*)(&Bs[0][r0 * 40 + c0]) = *(const uint4*)(W0 + k0);
    *(uint4*)(&Bs[0][r1 * 40 + c1]) = *(const uint4*)(W1 + k0);
    if (k0 + 32 < K) {
      __builtin_prefetch(A0 + k0 + 32, 0, 1);
      __builtin_prefetch(W0 + k0 + 32, 0, 1);
    }
    __syncthreads();

    v16bf afr[4], bfr[2];
#pragma unroll
    for (int r = 0; r < 4; ++r) {
      int idx = (waveM + r * 16 + mLane) * 40 + kOff;
      afr[r] = make_frag(*(const v8bf*)&As[0][idx], *(const v8bf*)&As[0][idx + 16]);
    }
#pragma unroll
    for (int c = 0; c < 2; ++c) {
      int idx = (waveN + c * 16 + mLane) * 40 + kOff;
      bfr[c] = make_frag(*(const v8bf*)&Bs[0][idx], *(const v8bf*)&Bs[0][idx + 16]);
    }
#pragma unroll
    for (int r = 0; r < 4; ++r)
#pragma unroll
      for (int c = 0; c < 2; ++c)
        acc[r][c] = __builtin_amdgcn_wmma_f32_16x16x32_bf16(
            false, afr[r], false, bfr[c], (short)0, acc[r][c], false, false);
    __syncthreads();
  }
#endif

#pragma unroll
  for (int r = 0; r < 4; ++r)
#pragma unroll
    for (int c = 0; c < 2; ++c) {
      int   n     = n0 + waveN + c * 16 + mLane;
      float bn    = bias[n];
      int   mBase = m0 + waveM + r * 16 + (lane >> 4) * 8;
#pragma unroll
      for (int q = 0; q < 8; ++q) {
        float  v   = acc[r][c][q] + bn;
        size_t off = (size_t)(mBase + q) * N + n;
        if (OUT_BF16) Cbf[off] = (bf16)v;
        else          Cf[off]  = v;
      }
    }
}

// ------------------------------------- att = softmax(scale*Q K^T + mask)
// One block per (b, h, 16 query rows). Raw scores in 64KB LDS, shfl softmax.
__global__ __launch_bounds__(256) void qk_softmax_kernel(
    const bf16* __restrict__ Qb, const bf16* __restrict__ Kb,
    const unsigned char* __restrict__ mask, float* __restrict__ attOut) {
  __shared__ float S[16 * 1024];

  const int t     = threadIdx.x;
  const int lane  = t & 31;
  const int w     = t >> 5;
  const int q0    = blockIdx.x * 16;
  const int h     = blockIdx.y;
  const int b     = blockIdx.z;
  const int mLane = lane & 15;
  const int kOff  = (lane >> 4) * 8;

  const bf16* qptr = Qb + (size_t)(b * NQS + q0 + mLane) * DDIM + h * HDEP;
  v16bf af0 = make_frag(*(const v8bf*)(qptr + kOff),      *(const v8bf*)(qptr + 16 + kOff));
  v16bf af1 = make_frag(*(const v8bf*)(qptr + 32 + kOff), *(const v8bf*)(qptr + 48 + kOff));

  const int n0w = w * 128;
#pragma unroll
  for (int t8 = 0; t8 < 8; ++t8) {
    const int   n0   = n0w + t8 * 16;
    const bf16* kptr = Kb + (size_t)(b * NKVS + n0 + mLane) * DDIM + h * HDEP;
    v16bf bf0 = make_frag(*(const v8bf*)(kptr + kOff),      *(const v8bf*)(kptr + 16 + kOff));
    v16bf bf1 = make_frag(*(const v8bf*)(kptr + 32 + kOff), *(const v8bf*)(kptr + 48 + kOff));
    v8f acc = vzero8();
    acc = __builtin_amdgcn_wmma_f32_16x16x32_bf16(false, af0, false, bf0, (short)0, acc, false, false);
    acc = __builtin_amdgcn_wmma_f32_16x16x32_bf16(false, af1, false, bf1, (short)0, acc, false, false);
#pragma unroll
    for (int q = 0; q < 8; ++q)
      S[((lane >> 4) * 8 + q) * 1024 + n0 + mLane] = acc[q];
  }
  __syncthreads();

  // row-wise softmax: thread (row = t>>4, seg = t&15) owns 64 columns.
  const int row = t >> 4;
  const int seg = t & 15;
  float* Srow = S + row * 1024 + seg * 64;
  const unsigned char* mrow =
      mask + ((size_t)b * NQS + q0 + row) * NKVS + seg * 64;

  float mx = -3.0e38f;
#pragma unroll 8
  for (int i = 0; i < 64; ++i) {
    float v = Srow[i] * ATT_SCALE;
    if (mrow[i]) v = -1.0e9f;
    Srow[i] = v;
    mx = fmaxf(mx, v);
  }
#pragma unroll
  for (int o = 1; o < 16; o <<= 1) mx = fmaxf(mx, __shfl_xor(mx, o, 16));

  float sum = 0.f;
#pragma unroll 8
  for (int i = 0; i < 64; ++i) sum += __expf(Srow[i] - mx);
#pragma unroll
  for (int o = 1; o < 16; o <<= 1) sum += __shfl_xor(sum, o, 16);
  const float rinv = 1.0f / sum;

  float* dst = attOut +
      (((size_t)(b * NH + h) * NQS) + q0 + row) * NKVS + seg * 64;
#pragma unroll
  for (int i = 0; i < 64; i += 4) {
    float4 o4;
    o4.x = __expf(Srow[i + 0] - mx) * rinv;
    o4.y = __expf(Srow[i + 1] - mx) * rinv;
    o4.z = __expf(Srow[i + 2] - mx) * rinv;
    o4.w = __expf(Srow[i + 3] - mx) * rinv;
    *(float4*)(dst + i) = o4;
  }
}

// -------------------------------------------------- out_1 = att @ V
// Block tile 128(M)x64(N) per (b,h); att re-read as f32, converted to bf16.
__global__ __launch_bounds__(256) void attv_kernel(
    const float* __restrict__ att, const bf16* __restrict__ Vbf,
    float* __restrict__ out1, bf16* __restrict__ o1tok) {
  __shared__ bf16 As[128 * 40];
  __shared__ bf16 Bs[64 * 40];

  const int t     = threadIdx.x;
  const int lane  = t & 31;
  const int w     = t >> 5;
  const int m0    = blockIdx.x * 128;
  const int h     = blockIdx.y;
  const int b     = blockIdx.z;
  const int waveM = w * 16;
  const int mLane = lane & 15;
  const int kOff  = (lane >> 4) * 8;

  const float* attBH = att + (size_t)(b * NH + h) * NQS * NKVS;

  v8f acc[4];
#pragma unroll
  for (int c = 0; c < 4; ++c) acc[c] = vzero8();

  for (int k0 = 0; k0 < NKVS; k0 += 32) {
    { // stage A: 128x32 f32 -> bf16; each thread converts 16 floats
      int row  = t >> 1;
      int colb = (t & 1) * 16;
      const float* srcA = attBH + (size_t)(m0 + row) * NKVS + k0 + colb;
      float4 f0 = *(const float4*)(srcA);
      float4 f1 = *(const float4*)(srcA + 4);
      float4 f2 = *(const float4*)(srcA + 8);
      float4 f3 = *(const float4*)(srcA + 12);
      *(v8bf*)&As[row * 40 + colb]     = cvt8(f0, f1);
      *(v8bf*)&As[row * 40 + colb + 8] = cvt8(f2, f3);
    }
    { // stage B: V[k0..k0+31, h*64..+63] transposed -> Bs[n][k]
      int kk = t >> 3;
      int nb = (t & 7) * 8;
      v8bf v = *(const v8bf*)(Vbf + (size_t)(b * NKVS + k0 + kk) * DDIM + h * HDEP + nb);
#pragma unroll
      for (int e = 0; e < 8; ++e) Bs[(nb + e) * 40 + kk] = v[e];
    }
    __syncthreads();

    int   ia = (waveM + mLane) * 40 + kOff;
    v16bf af = make_frag(*(const v8bf*)&As[ia], *(const v8bf*)&As[ia + 16]);
#pragma unroll
    for (int c = 0; c < 4; ++c) {
      int   ib = (c * 16 + mLane) * 40 + kOff;
      v16bf bf_ = make_frag(*(const v8bf*)&Bs[ib], *(const v8bf*)&Bs[ib + 16]);
      acc[c] = __builtin_amdgcn_wmma_f32_16x16x32_bf16(
          false, af, false, bf_, (short)0, acc[c], false, false);
    }
    __syncthreads();
  }

#pragma unroll
  for (int c = 0; c < 4; ++c) {
    int n = c * 16 + mLane;
#pragma unroll
    for (int q = 0; q < 8; ++q) {
      int   m = m0 + waveM + (lane >> 4) * 8 + q;
      float v = acc[c][q];
      out1[((size_t)(b * NH + h) * NQS + m) * HDEP + n] = v;
      o1tok[((size_t)(b * NQS) + m) * DDIM + h * HDEP + n] = (bf16)v;
    }
  }
}

// ------------------------------------------------------------------ launch
extern "C" void kernel_launch(void* const* d_in, const int* in_sizes, int n_in,
                              void* d_out, int out_size, void* d_ws, size_t ws_size,
                              hipStream_t stream) {
  (void)in_sizes; (void)n_in; (void)out_size; (void)ws_size;

  const float*         V_token  = (const float*)d_in[0];
  const float*         L_token  = (const float*)d_in[1];
  const unsigned char* pad_mask = (const unsigned char*)d_in[2];
  const float* Wq_w = (const float*)d_in[3];
  const float* Wq_b = (const float*)d_in[4];
  const float* Wk_w = (const float*)d_in[5];
  const float* Wk_b = (const float*)d_in[6];
  const float* Wv_w = (const float*)d_in[7];
  const float* Wv_b = (const float*)d_in[8];
  const float* Wo_w = (const float*)d_in[9];
  const float* Wo_b = (const float*)d_in[10];

  float* out1 = (float*)d_out;                       // [B,H,NQ,DEPTH]
  float* outp = out1 + (size_t)BB * NH * NQS * HDEP; // [B,NQ,D]
  float* attp = outp + (size_t)BB * NQS * DDIM;      // [B,H,NQ,NKV]

  const size_t nTok = (size_t)BB * NQS * DDIM;   // 4M elems
  const size_t nW   = (size_t)DDIM * DDIM;       // 1M elems
  bf16* Vt  = (bf16*)d_ws;
  bf16* Lt  = Vt  + nTok;
  bf16* Wqb = Lt  + nTok;
  bf16* Wkb = Wqb + nW;
  bf16* Wvb = Wkb + nW;
  bf16* Wob = Wvb + nW;
  bf16* Qbf = Wob + nW;
  bf16* Kbf = Qbf + nTok;
  bf16* Vbf = Kbf + nTok;
  bf16* O1t = Vbf + nTok;

  // 1) fp32 -> bf16 conversions
  {
    int n8 = (int)(nTok / 8);
    cvt_f32_bf16_kernel<<<(n8 + 255) / 256, 256, 0, stream>>>(V_token, Vt, n8);
    cvt_f32_bf16_kernel<<<(n8 + 255) / 256, 256, 0, stream>>>(L_token, Lt, n8);
    int w8 = (int)(nW / 8);
    cvt_f32_bf16_kernel<<<(w8 + 255) / 256, 256, 0, stream>>>(Wq_w, Wqb, w8);
    cvt_f32_bf16_kernel<<<(w8 + 255) / 256, 256, 0, stream>>>(Wk_w, Wkb, w8);
    cvt_f32_bf16_kernel<<<(w8 + 255) / 256, 256, 0, stream>>>(Wv_w, Wvb, w8);
    cvt_f32_bf16_kernel<<<(w8 + 255) / 256, 256, 0, stream>>>(Wo_w, Wob, w8);
  }

  // 2) Q/K/V projections (x @ W^T + b), bf16 outputs
  dim3 gp(DDIM / 128, (BB * NQS) / 128);
  gemm_xt_bias<true><<<gp, 256, 0, stream>>>(Vt, Wqb, Wq_b, Qbf, nullptr, BB * NQS, DDIM, DDIM);
  gemm_xt_bias<true><<<gp, 256, 0, stream>>>(Lt, Wkb, Wk_b, Kbf, nullptr, BB * NQS, DDIM, DDIM);
  gemm_xt_bias<true><<<gp, 256, 0, stream>>>(Lt, Wvb, Wv_b, Vbf, nullptr, BB * NQS, DDIM, DDIM);

  // 3) att = softmax(scale * Q K^T, mask)  (writes f32 att output)
  qk_softmax_kernel<<<dim3(NQS / 16, NH, BB), 256, 0, stream>>>(Qbf, Kbf, pad_mask, attp);

  // 4) out_1 = att @ V  (f32 output + bf16 token-layout copy)
  attv_kernel<<<dim3(NQS / 128, NH, BB), 256, 0, stream>>>(attp, Vbf, out1, O1t);

  // 5) out = out_1 @ Wo^T + b
  gemm_xt_bias<false><<<gp, 256, 0, stream>>>(O1t, Wob, Wo_b, nullptr, outp, BB * NQS, DDIM, DDIM);
}
